// Model_74
// MI455X (gfx1250) — matrix operand order repaired and checked by reading, not compiled, not run on hardware
//
#include <hip/hip_runtime.h>
#include <hip/hip_bf16.h>
#include <utility>

typedef __attribute__((ext_vector_type(16))) _Float16 v16h;
typedef __attribute__((ext_vector_type(8)))  float    v8f;

#define D64 64

// ---------------- generic ----------------
__global__ void zero_f32(float* p, long n) {
  long t = (long)blockIdx.x * blockDim.x + threadIdx.x;
  if (t < n) p[t] = 0.f;
}

__global__ void axpy_kernel(float* acc, const float* x, long n) {
  long t = (long)blockIdx.x * blockDim.x + threadIdx.x;
  if (t < n) acc[t] += x[t];
}

// ---------------- graph normalization ----------------
__global__ void degree_kernel(const int* row, const int* col, float* dr, float* dc, long E) {
  long e = (long)blockIdx.x * blockDim.x + threadIdx.x;
  if (e >= E) return;
  atomicAdd(&dr[row[e]], 1.f);
  atomicAdd(&dc[col[e]], 1.f);
}

__global__ void finalize_deg(float* d, int N) {
  int n = blockIdx.x * blockDim.x + threadIdx.x;
  if (n >= N) return;
  float v = d[n];
  d[n] = (v > 0.f) ? rsqrtf(fmaxf(v, 1.f)) : 0.f;
}

__global__ void vals_kernel(float* vals, const float* dr, const float* dc,
                            const int* row, const int* col, long E) {
  long e = (long)blockIdx.x * blockDim.x + threadIdx.x;
  if (e >= E) return;
  vals[e] = dr[row[e]] * dc[col[e]];
}

// ---------------- SpMM (atomic scatter), optional edge/node gates ----------------
__global__ void spmm_atomic(float* out, const float* x, const int* row, const int* col,
                            const float* vals, const float* egate, const float* ngate, long E) {
  long t = (long)blockIdx.x * blockDim.x + threadIdx.x;
  long total = E << 4;             // 16 chunks of 4 dims
  if (t >= total) return;
  int  chunk = (int)(t & 15);
  long e     = t >> 4;
  int r = row[e], c = col[e];
  float v = vals[e];
  if (egate) v *= egate[e];
  if (ngate) v *= ngate[c];
  long e2 = e + 512;
  if (e2 < E) __builtin_prefetch(x + (long)col[e2] * D64 + chunk * 4, 0, 1);
  const float4 xv = *(const float4*)(x + (long)c * D64 + chunk * 4);
  float* o = out + (long)r * D64 + chunk * 4;
  atomicAdd(o + 0, v * xv.x);
  atomicAdd(o + 1, v * xv.y);
  atomicAdd(o + 2, v * xv.z);
  atomicAdd(o + 3, v * xv.w);
}

// ---------------- gate projections ----------------
__global__ void edge_proj(const float* base, const float* w /*2*64*/, float* prA, float* prB, int N) {
  __shared__ float sw[128];
  if (threadIdx.x < 128) sw[threadIdx.x] = w[threadIdx.x];
  __syncthreads();
  int n = blockIdx.x * blockDim.x + threadIdx.x;
  if (n >= N) return;
  const float* xr = base + (long)n * D64;
  float a = 0.f, b = 0.f;
  for (int d = 0; d < D64; ++d) { float x = xr[d]; a += x * sw[d]; b += x * sw[64 + d]; }
  prA[n] = a; prB[n] = b;
}

__global__ void edge_gate(const float* prA, const float* prB, const int* row, const int* col,
                          const float* eb, int l, const float* noise, float* mbuf,
                          float* slot, long E, float invfac) {
  long e = (long)blockIdx.x * blockDim.x + threadIdx.x;
  float m = 0.f;
  if (e < E) {
    float logit = prA[row[e]] + prB[col[e]] + eb[l];
    m = 1.f / (1.f + __expf(-(logit + noise[e]) * 5.0f));   // /SSL_T = *5
    mbuf[e] = m;
  }
  for (int off = 16; off > 0; off >>= 1) m += __shfl_xor(m, off, 32);
  if ((threadIdx.x & 31) == 0) atomicAdd(slot, m * invfac);
}

__global__ void node_gate(const float* base, const float* w /*64*/, const float* nb, int l,
                          const float* noise, float* nm, float* slot, int N, float invfac) {
  __shared__ float sw[64];
  if (threadIdx.x < 64) sw[threadIdx.x] = w[threadIdx.x];
  __syncthreads();
  int n = blockIdx.x * blockDim.x + threadIdx.x;
  float m = 0.f;
  if (n < N) {
    const float* xr = base + (long)n * D64;
    float a = 0.f;
    for (int d = 0; d < D64; ++d) a += xr[d] * sw[d];
    m = 1.f / (1.f + __expf(-(a + nb[l] + noise[n]) * 5.0f));
    nm[n] = m;
  }
  for (int off = 16; off > 0; off >>= 1) m += __shfl_xor(m, off, 32);
  if ((threadIdx.x & 31) == 0) atomicAdd(slot, m * invfac);
}

// ---------------- gather / bpr / l2norm / f16 ----------------
__global__ void gather_rows(const float* src, const int* idx, int offset, float scale,
                            float* dst, int B) {
  int t = blockIdx.x * blockDim.x + threadIdx.x;
  if (t >= B * D64) return;
  int b = t >> 6, d = t & 63;
  dst[t] = src[(long)(idx[b] + offset) * D64 + d] * scale;
}

__global__ void bpr_kernel(const float* u, const float* p, const float* n, float* slot, int B) {
  int gid = blockIdx.x * blockDim.x + threadIdx.x;
  int w = gid >> 5, lane = gid & 31;
  if (w >= B) return;
  const float* ur = u + (long)w * D64;
  const float* pr = p + (long)w * D64;
  const float* nr = n + (long)w * D64;
  float dp = 0.f, dn = 0.f;
  for (int k = 0; k < 2; ++k) {
    int d = lane + 32 * k;
    float uv = ur[d];
    dp += uv * pr[d]; dn += uv * nr[d];
  }
  for (int off = 16; off > 0; off >>= 1) { dp += __shfl_xor(dp, off, 32); dn += __shfl_xor(dn, off, 32); }
  if (lane == 0) {
    float x = dp - dn;
    float ls = (x >= 0.f) ? -log1pf(__expf(-x)) : (x - log1pf(__expf(x)));
    atomicAdd(slot, -ls / (float)B);
  }
}

__global__ void l2norm_kernel(const float* src, float* dst, int B) {
  int gid = blockIdx.x * blockDim.x + threadIdx.x;
  int w = gid >> 5, lane = gid & 31;
  if (w >= B) return;
  const float* xr = src + (long)w * D64;
  float s = 0.f;
  for (int k = 0; k < 2; ++k) { float v = xr[lane + 32 * k]; s += v * v; }
  for (int off = 16; off > 0; off >>= 1) s += __shfl_xor(s, off, 32);
  float inv = 1.f / fmaxf(sqrtf(s), 1e-12f);
  for (int k = 0; k < 2; ++k) { int d = lane + 32 * k; dst[(long)w * D64 + d] = xr[d] * inv; }
}

__global__ void f2h_kernel(const float* src, _Float16* dst, int n) {
  int t = blockIdx.x * blockDim.x + threadIdx.x;
  if (t < n) dst[t] = (_Float16)src[t];
}

// ---------------- WMMA similarity: sim = exp((z1 z2^T)/SSL_T) ----------------
// 8 waves/block; block stages a 256-column z2 panel (32 KB) into LDS via CDNA5
// async global->LDS (ASYNCcnt), then each wave computes 2 16x16 tiles with
// A-fragment reuse, B-fragments from LDS via ds_load_b128.
#define SIMPANEL 256
__global__ void sim_wmma(const _Float16* __restrict__ z1, const _Float16* __restrict__ z2,
                         float* __restrict__ sim, int B) {
  __shared__ _Float16 lds[SIMPANEL * D64];   // 32 KB
  const int tid  = threadIdx.x;
  const int lane = tid & 31;
  const int wave = tid >> 5;
  const int ti     = blockIdx.x * 16;          // output row tile
  const int jpanel = blockIdx.y * SIMPANEL;    // panel of output cols

  // --- async stage: z2[jpanel .. jpanel+256) rows (64 halves each) -> LDS ---
  unsigned lbase = (unsigned)(size_t)(&lds[0]);
  unsigned long gbase = (unsigned long)(const void*)(z2 + (size_t)jpanel * D64);
  #pragma unroll
  for (int c = 0; c < 8; ++c) {               // 2048 16B-chunks / 256 threads
    int chunk = tid + c * 256;
    unsigned      ldsoff = lbase + (unsigned)chunk * 16u;
    unsigned long gaddr  = gbase + (unsigned long)chunk * 16ul;
    asm volatile("global_load_async_to_lds_b128 %0, %1, off"
                 :: "v"(ldsoff), "v"(gaddr) : "memory");
  }
  asm volatile("s_wait_asynccnt 0x0" ::: "memory");
  __syncthreads();

  const int m    = lane & 15;
  const int half = lane >> 4;
  const int jw   = wave * 32;                 // this wave's 32 columns in panel

  v8f c0 = {}, c1 = {};
  #pragma unroll
  for (int k0 = 0; k0 < D64; k0 += 32) {
    // A fragment: lane m=M; halves [8*half .. +7] and [16+8*half .. +7]
    v16h a;
    const _Float16* arow = z1 + (size_t)(ti + m) * D64 + k0 + 8 * half;
    ((float4*)&a)[0] = *(const float4*)(arow);
    ((float4*)&a)[1] = *(const float4*)(arow + 16);
    // B fragments from LDS: lane n=N; halves [8*half .. +7] and [16+8*half .. +7] (B[k][n] = z2[j+n][k])
    v16h b0, b1;
    const _Float16* br0 = &lds[(jw +      m) * D64 + k0 + 8 * half];
    const _Float16* br1 = &lds[(jw + 16 + m) * D64 + k0 + 8 * half];
    ((float4*)&b0)[0] = *(const float4*)(br0);
    ((float4*)&b0)[1] = *(const float4*)(br0 + 16);
    ((float4*)&b1)[0] = *(const float4*)(br1);
    ((float4*)&b1)[1] = *(const float4*)(br1 + 16);
    c0 = __builtin_amdgcn_wmma_f32_16x16x32_f16(false, a, false, b0, (short)0, c0, false, false);
    c1 = __builtin_amdgcn_wmma_f32_16x16x32_f16(false, a, false, b1, (short)0, c1, false, false);
  }
  // C/D layout: VGPR v -> M = v + 8*half, N = lane%16
  #pragma unroll
  for (int v = 0; v < 8; ++v) {
    int mi = v + 8 * half;
    sim[(size_t)(ti + mi) * B + jpanel + jw +      m] = __expf(c0[v] * 5.0f);  // exp(dot/0.2)
    sim[(size_t)(ti + mi) * B + jpanel + jw + 16 + m] = __expf(c1[v] * 5.0f);
  }
}

// ---------------- per-row top-K threshold via binary search (row in LDS) ----------------
__global__ void topk_thresh(const float* sim, float* thresh, int B, int K) {
  __shared__ float srow[2048];
  __shared__ float sred[256];
  __shared__ int   ired[256];
  int i = blockIdx.x, t = threadIdx.x;
  for (int j = t; j < B; j += 256) srow[j] = sim[(long)i * B + j];
  __syncthreads();
  float mn = 1e30f, mx = -1e30f;
  for (int j = t; j < B; j += 256) { float v = srow[j]; mn = fminf(mn, v); mx = fmaxf(mx, v); }
  sred[t] = mx; __syncthreads();
  for (int s = 128; s > 0; s >>= 1) { if (t < s) sred[t] = fmaxf(sred[t], sred[t + s]); __syncthreads(); }
  mx = sred[0]; __syncthreads();
  sred[t] = mn; __syncthreads();
  for (int s = 128; s > 0; s >>= 1) { if (t < s) sred[t] = fminf(sred[t], sred[t + s]); __syncthreads(); }
  mn = sred[0]; __syncthreads();
  float lo = mn, hi = mx;
  for (int it = 0; it < 32; ++it) {
    float mid = 0.5f * (lo + hi);
    int c = 0;
    for (int j = t; j < B; j += 256) c += (srow[j] >= mid) ? 1 : 0;
    ired[t] = c; __syncthreads();
    for (int s = 128; s > 0; s >>= 1) { if (t < s) ired[t] += ired[t + s]; __syncthreads(); }
    int cnt = ired[0]; __syncthreads();
    if (cnt >= K) lo = mid; else hi = mid;
  }
  if (t == 0) thresh[i] = lo;
}

// ---------------- mlp projections a_i = z1.mlp_w[:D], b_j = z2.mlp_w[D:] ----------------
__global__ void ab_proj(const float* z1, const float* z2, const float* mlp_w,
                        float* av, float* bv, int B) {
  __shared__ float sw[128];
  if (threadIdx.x < 128) sw[threadIdx.x] = mlp_w[threadIdx.x];
  __syncthreads();
  int i = blockIdx.x * blockDim.x + threadIdx.x;
  if (i >= B) return;
  float a = 0.f, b = 0.f;
  for (int d = 0; d < D64; ++d) {
    a += z1[(long)i * D64 + d] * sw[d];
    b += z2[(long)i * D64 + d] * sw[64 + d];
  }
  av[i] = a; bv[i] = b;
}

// ---------------- SSL score: gated top-K positive aggregation ----------------
__global__ void ssl_score(const float* sim, const float* thresh, const float* av, const float* bv,
                          const float* noise, const float* z1, const float* z2,
                          const float* mlp_b, float* slot, int B) {
  __shared__ float red[256];
  int i = blockIdx.x, t = threadIdx.x;
  float th = thresh[i];
  float ai = av[i] + mlp_b[0];
  float part = 0.f;
  for (int j = t; j < B; j += 256) {
    float s = sim[(long)i * B + j];
    if (j != i && s >= th) {
      // gate = sigmoid((mlp+noise)/CH_T) > TAO=0.5  <=>  mlp+noise > 0
      if (ai + bv[j] + noise[(long)i * B + j] > 0.f) part += s;
    }
  }
  red[t] = part; __syncthreads();
  for (int s = 128; s > 0; s >>= 1) { if (t < s) red[t] += red[t + s]; __syncthreads(); }
  if (t == 0) {
    float diag = 0.f;
    for (int d = 0; d < D64; ++d) diag += z1[(long)i * D64 + d] * z2[(long)i * D64 + d];
    float v = red[0] + diag;
    atomicAdd(slot, -logf(fmaxf(v, 1e-8f)) / (float)B);
  }
}

// ---------------- final combine ----------------
__global__ void finalize(const float* s, float* out) {
  // s: 0=rec 1=rec_e 2=rec_n 3=ereg(/L done) 4=nreg(/L done) 5=cl
  out[0] = s[0] + 0.1f * (s[1] + s[2]) + 0.1f * s[5] + 0.01f * (s[3] + s[4]);
}

// =======================================================================
extern "C" void kernel_launch(void* const* d_in, const int* in_sizes, int n_in,
                              void* d_out, int out_size, void* d_ws, size_t ws_size,
                              hipStream_t stream) {
  (void)in_sizes; (void)n_in; (void)out_size; (void)ws_size;
  const int  U_ = 100000, I_ = 50000, N_ = 150000, B_ = 2048;
  const long E_ = 1600000;
  const long NB = (long)N_ * D64;           // 9.6M floats
  const int  K_ = (int)(0.05f * B_);        // 102

  const int*   user_id    = (const int*)d_in[0];
  const int*   pos_item   = (const int*)d_in[1];
  const int*   neg_item   = (const int*)d_in[2];
  const int*   row        = (const int*)d_in[3];
  const int*   col        = (const int*)d_in[4];
  const float* user_emb   = (const float*)d_in[5];
  const float* item_emb   = (const float*)d_in[6];
  const float* edge_w     = (const float*)d_in[7];
  const float* edge_b     = (const float*)d_in[8];
  const float* node_w     = (const float*)d_in[9];
  const float* node_b     = (const float*)d_in[10];
  const float* mlp_w      = (const float*)d_in[11];
  const float* mlp_b      = (const float*)d_in[12];
  const float* edge_noise = (const float*)d_in[13];
  const float* node_noise = (const float*)d_in[14];
  const float* judge_noise= (const float*)d_in[15];

  // ---- workspace bump allocator ----
  char* wsp = (char*)d_ws;
  size_t off = 0;
  auto alloc = [&](size_t bytes) -> char* {
    char* p = wsp + off;
    off += (bytes + 255) & ~(size_t)255;
    return p;
  };
  float* emb0 = (float*)alloc(NB * 4);
  float* bigA = (float*)alloc(NB * 4);
  float* bigB = (float*)alloc(NB * 4);   // running accumulator (all chains)
  float* bigC = (float*)alloc(NB * 4);
  float* bigD = (float*)alloc(NB * 4);
  float* degr = (float*)alloc((size_t)N_ * 4);
  float* degc = (float*)alloc((size_t)N_ * 4);
  float* vals = (float*)alloc((size_t)E_ * 4);
  float* mbuf = (float*)alloc((size_t)E_ * 4);
  float* prA  = (float*)alloc((size_t)N_ * 4);
  float* prB  = (float*)alloc((size_t)N_ * 4);
  float* nm   = (float*)alloc((size_t)N_ * 4);
  float* g[9];
  for (int i = 0; i < 9; ++i) g[i] = (float*)alloc((size_t)B_ * D64 * 4);
  float* z[6];
  for (int i = 0; i < 6; ++i) z[i] = (float*)alloc((size_t)B_ * D64 * 4);
  _Float16* h[6];
  for (int i = 0; i < 6; ++i) h[i] = (_Float16*)alloc((size_t)B_ * D64 * 2);
  float* sim  = (float*)alloc((size_t)B_ * B_ * 4);
  float* thr  = (float*)alloc((size_t)B_ * 4);
  float* av   = (float*)alloc((size_t)B_ * 4);
  float* bv   = (float*)alloc((size_t)B_ * 4);
  float* scal = (float*)alloc(16 * 4);

  const int blkNB  = (int)((NB + 255) / 256);
  const int blkE16 = (int)((E_ * 16 + 255) / 256);
  const int blkE   = (int)((E_ + 255) / 256);
  const int blkN   = (N_ + 255) / 256;
  const int blkBD  = (B_ * D64 + 255) / 256;
  const int blkW   = (B_ * 32 + 255) / 256;

  // ---- emb0 = concat(user_emb, item_emb); zero accumulator slots + degrees ----
  hipMemcpyAsync(emb0, user_emb, (size_t)U_ * D64 * 4, hipMemcpyDeviceToDevice, stream);
  hipMemcpyAsync(emb0 + (long)U_ * D64, item_emb, (size_t)I_ * D64 * 4, hipMemcpyDeviceToDevice, stream);
  zero_f32<<<1, 16, 0, stream>>>(scal, 16);
  zero_f32<<<blkN, 256, 0, stream>>>(degr, N_);
  zero_f32<<<blkN, 256, 0, stream>>>(degc, N_);
  degree_kernel<<<blkE, 256, 0, stream>>>(row, col, degr, degc, E_);
  finalize_deg<<<blkN, 256, 0, stream>>>(degr, N_);
  finalize_deg<<<blkN, 256, 0, stream>>>(degc, N_);
  vals_kernel<<<blkE, 256, 0, stream>>>(vals, degr, degc, row, col, E_);

  // ---- chain 1: plain LightGCN propagation ----
  float* cur = bigA; float* tmp = bigC;
  hipMemcpyAsync(cur,  emb0, NB * 4, hipMemcpyDeviceToDevice, stream);
  hipMemcpyAsync(bigB, emb0, NB * 4, hipMemcpyDeviceToDevice, stream);
  for (int l = 0; l < 3; ++l) {
    zero_f32<<<blkNB, 256, 0, stream>>>(tmp, NB);
    spmm_atomic<<<blkE16, 256, 0, stream>>>(tmp, cur, row, col, vals, nullptr, nullptr, E_);
    std::swap(cur, tmp);
    axpy_kernel<<<blkNB, 256, 0, stream>>>(bigB, cur, NB);
  }
  float* base = cur;            // == bigC after 3 swaps; kept through chains 2 & 3
  float* freeb = tmp;           // == bigA
  const float scaleL = 1.f / 4.f;
  gather_rows<<<blkBD, 256, 0, stream>>>(bigB, user_id,  0,  scaleL, g[0], B_);
  gather_rows<<<blkBD, 256, 0, stream>>>(bigB, pos_item, U_, scaleL, g[1], B_);
  gather_rows<<<blkBD, 256, 0, stream>>>(bigB, neg_item, U_, scaleL, g[2], B_);

  // ---- chain 2: edge-gated propagation ----
  float* cur2 = bigD; float* tmp2 = freeb;
  hipMemcpyAsync(cur2, emb0, NB * 4, hipMemcpyDeviceToDevice, stream);
  hipMemcpyAsync(bigB, emb0, NB * 4, hipMemcpyDeviceToDevice, stream);
  const float invEL = 1.f / ((float)E_ * 3.f);
  for (int l = 0; l < 3; ++l) {
    edge_proj<<<blkN, 256, 0, stream>>>(base, edge_w + (long)l * 128, prA, prB, N_);
    edge_gate<<<blkE, 256, 0, stream>>>(prA, prB, row, col, edge_b, l,
                                        edge_noise + (long)l * E_, mbuf, scal + 3, E_, invEL);
    zero_f32<<<blkNB, 256, 0, stream>>>(tmp2, NB);
    spmm_atomic<<<blkE16, 256, 0, stream>>>(tmp2, cur2, row, col, vals, mbuf, nullptr, E_);
    std::swap(cur2, tmp2);
    axpy_kernel<<<blkNB, 256, 0, stream>>>(bigB, cur2, NB);
  }
  gather_rows<<<blkBD, 256, 0, stream>>>(bigB, user_id,  0,  scaleL, g[3], B_);
  gather_rows<<<blkBD, 256, 0, stream>>>(bigB, pos_item, U_, scaleL, g[4], B_);
  gather_rows<<<blkBD, 256, 0, stream>>>(bigB, neg_item, U_, scaleL, g[5], B_);

  // ---- chain 3: node-gated propagation ----
  float* cur3 = tmp2; float* tmp3 = cur2;   // reuse freed ping-pong buffers
  hipMemcpyAsync(cur3, emb0, NB * 4, hipMemcpyDeviceToDevice, stream);
  hipMemcpyAsync(bigB, emb0, NB * 4, hipMemcpyDeviceToDevice, stream);
  const float invNL = 1.f / ((float)N_ * 3.f);
  for (int l = 0; l < 3; ++l) {
    node_gate<<<blkN, 256, 0, stream>>>(base, node_w + (long)l * D64, node_b, l,
                                        node_noise + (long)l * N_, nm, scal + 4, N_, invNL);
    zero_f32<<<blkNB, 256, 0, stream>>>(tmp3, NB);
    spmm_atomic<<<blkE16, 256, 0, stream>>>(tmp3, cur3, row, col, vals, nullptr, nm, E_);
    std::swap(cur3, tmp3);
    axpy_kernel<<<blkNB, 256, 0, stream>>>(bigB, cur3, NB);
  }
  gather_rows<<<blkBD, 256, 0, stream>>>(bigB, user_id,  0,  scaleL, g[6], B_);
  gather_rows<<<blkBD, 256, 0, stream>>>(bigB, pos_item, U_, scaleL, g[7], B_);
  gather_rows<<<blkBD, 256, 0, stream>>>(bigB, neg_item, U_, scaleL, g[8], B_);

  // ---- BPR losses ----
  bpr_kernel<<<blkW, 256, 0, stream>>>(g[0], g[1], g[2], scal + 0, B_);
  bpr_kernel<<<blkW, 256, 0, stream>>>(g[3], g[4], g[5], scal + 1, B_);
  bpr_kernel<<<blkW, 256, 0, stream>>>(g[6], g[7], g[8], scal + 2, B_);

  // ---- normalize: nu ni nue nie nun nin ----
  const float* nsrc[6] = { g[0], g[1], g[3], g[4], g[6], g[7] };
  for (int i = 0; i < 6; ++i) {
    l2norm_kernel<<<blkW, 256, 0, stream>>>(nsrc[i], z[i], B_);
    f2h_kernel<<<blkBD, 256, 0, stream>>>(z[i], h[i], B_ * D64);
  }

  // ---- 4 SSL terms: (nue,nu) (nie,ni) (nun,nu) (nin,ni) ----
  const int z1i[4] = { 2, 3, 4, 5 };
  const int z2i[4] = { 0, 1, 0, 1 };
  dim3 simGrid(B_ / 16, B_ / SIMPANEL);   // 128 x 8
  for (int t = 0; t < 4; ++t) {
    sim_wmma<<<simGrid, 256, 0, stream>>>(h[z1i[t]], h[z2i[t]], sim, B_);
    topk_thresh<<<B_, 256, 0, stream>>>(sim, thr, B_, K_);
    ab_proj<<<(B_ + 255) / 256, 256, 0, stream>>>(z[z1i[t]], z[z2i[t]], mlp_w, av, bv, B_);
    ssl_score<<<B_, 256, 0, stream>>>(sim, thr, av, bv,
                                      judge_noise + (long)t * B_ * B_,
                                      z[z1i[t]], z[z2i[t]], mlp_b, scal + 5, B_);
  }

  finalize<<<1, 1, 0, stream>>>(scal, (float*)d_out);
}
